// ConformerSelfAttention_52991306498603
// MI455X (gfx1250) — matrix operand order repaired and checked by reading, not compiled, not run on hardware
//
#include <hip/hip_runtime.h>

typedef __attribute__((ext_vector_type(16))) __bf16 bf16x16;
typedef __attribute__((ext_vector_type(8)))  __bf16 bf16x8;
typedef __attribute__((ext_vector_type(8)))  float  f32x8;

#define B_  8
#define T_  1024
#define D_  512
#define H_  8
#define DK_ 64

// ---- CDNA5 async global->LDS path (ASYNCcnt), with portable fallback ----
#if __has_builtin(__builtin_amdgcn_global_load_async_to_lds_b128)
#define HAVE_ASYNC_LDS 1
typedef int v4i_vs __attribute__((vector_size(16)));   // matches builtin param type
#endif

static __device__ __forceinline__ void async_copy16(const __bf16* g, __bf16* l) {
#ifdef HAVE_ASYNC_LDS
  __builtin_amdgcn_global_load_async_to_lds_b128((v4i_vs*)g, (v4i_vs*)l, 0, 0);
#else
  *(bf16x8*)l = *(const bf16x8*)g;
#endif
}

static __device__ __forceinline__ void wait_async_all() {
#ifdef HAVE_ASYNC_LDS
#if __has_builtin(__builtin_amdgcn_s_wait_asynccnt)
  __builtin_amdgcn_s_wait_asynccnt(0);
#else
  asm volatile("s_wait_asynccnt 0" ::: "memory");
#endif
#endif
}

static __device__ __forceinline__ f32x8 wmma_bf16(bf16x16 a, bf16x16 b, f32x8 c) {
  return __builtin_amdgcn_wmma_f32_16x16x32_bf16(false, a, false, b, (short)0, c, false, false);
}

// A fragment (16x32, bf16): lane m=L&15, half h=L>>4.
// elements 0..7 -> K = 8h+e ; elements 8..15 -> K = 16+8h+(e-8)  (ISA 7.12.2)
static __device__ __forceinline__ bf16x16 frag_a(const __bf16* base, int lda, int lane) {
  int m = lane & 15, hl = lane >> 4;
  const __bf16* r0 = base + (size_t)m * lda + 8 * hl;
  bf16x8 lo = *(const bf16x8*)(r0);
  bf16x8 hi = *(const bf16x8*)(r0 + 16);
  return __builtin_shufflevector(lo, hi, 0,1,2,3,4,5,6,7,8,9,10,11,12,13,14,15);
}

// B fragment (32x16, bf16) from an n-major tile (row = n, col = k):
// lane n=L&15, half h=L>>4; elements 0..7 -> K = 8h+e ; elements 8..15 -> K = 16+8h+(e-8)
static __device__ __forceinline__ bf16x16 frag_b(const __bf16* base, int ldb, int lane) {
  int n = lane & 15, hl = lane >> 4;
  const __bf16* r0 = base + (size_t)n * ldb + 8 * hl;
  bf16x8 lo = *(const bf16x8*)(r0);
  bf16x8 hi = *(const bf16x8*)(r0 + 16);
  return __builtin_shufflevector(lo, hi, 0,1,2,3,4,5,6,7,8,9,10,11,12,13,14,15);
}

// ---------------- LayerNorm: one wave per row, fp32 in -> bf16 out ----------------
__global__ __launch_bounds__(256) void ln_kernel(const float* __restrict__ x,
                                                 const float* __restrict__ gamma,
                                                 const float* __restrict__ beta,
                                                 __bf16* __restrict__ xn) {
  int wave = threadIdx.x >> 5, lane = threadIdx.x & 31;
  int row  = blockIdx.x * 8 + wave;                 // 1024 blocks * 8 waves = 8192 rows
  const float* xr = x + (size_t)row * D_;
  float v[16], s = 0.f, s2 = 0.f;
#pragma unroll
  for (int i = 0; i < 16; ++i) { v[i] = xr[lane + 32 * i]; s += v[i]; s2 += v[i] * v[i]; }
#pragma unroll
  for (int off = 16; off; off >>= 1) { s += __shfl_xor(s, off, 32); s2 += __shfl_xor(s2, off, 32); }
  float mean = s * (1.f / D_);
  float var  = s2 * (1.f / D_) - mean * mean;
  float inv  = rsqrtf(var + 1e-5f);
  __bf16* orow = xn + (size_t)row * D_;
#pragma unroll
  for (int i = 0; i < 16; ++i) {
    int c = lane + 32 * i;
    orow[c] = (__bf16)((v[i] - mean) * inv * gamma[c] + beta[c]);
  }
}

// ---------------- weight fp32 (K x N) -> bf16 transposed (N x K) ----------------
__global__ __launch_bounds__(256) void transpose_bf16_kernel(const float* __restrict__ W,
                                                             __bf16* __restrict__ Wt,
                                                             int K, int N) {
  int idx = blockIdx.x * 256 + threadIdx.x;
  if (idx < K * N) {
    int k = idx / N, n = idx % N;
    Wt[(size_t)n * K + k] = (__bf16)W[idx];
  }
}

__global__ __launch_bounds__(256) void convert_bf16_kernel(const float* __restrict__ src,
                                                           __bf16* __restrict__ dst, int n) {
  int idx = blockIdx.x * 256 + threadIdx.x;
  if (idx < n) dst[idx] = (__bf16)src[idx];
}

// stage one 128x32 A-tile + 128x32 B-tile into LDS (async on CDNA5)
static __device__ __forceinline__ void stage_gemm_tile(const __bf16* __restrict__ A,
                                                       const __bf16* __restrict__ Bt,
                                                       __bf16* sA, __bf16* sB, int tid,
                                                       int row0, int col0, int M, int N,
                                                       int K, int k0) {
#pragma unroll
  for (int i = 0; i < 2; ++i) {
    int c = tid + 256 * i;
    int r = c >> 2, cc = (c & 3) * 8;
    int gr = row0 + r; gr = gr < M ? gr : M - 1;
    async_copy16(&A[(size_t)gr * K + k0 + cc], &sA[r * 32 + cc]);
    int gn = col0 + r; gn = gn < N ? gn : N - 1;
    async_copy16(&Bt[(size_t)gn * K + k0 + cc], &sB[r * 32 + cc]);
  }
}

// ---------------- tiled WMMA GEMM: C = A(MxK) * Bt(NxK)^T, 128x128 tile ----------------
// double-buffered LDS; out0 = acc+bias0+bias_u (bf16), out1 = acc+bias0+bias_v (bf16),
// outf = acc+bias0 (f32)
__global__ __launch_bounds__(256) void gemm_bf16_kernel(const __bf16* __restrict__ A,
                                                        const __bf16* __restrict__ Bt,
                                                        const float* bias0, const float* bias_u,
                                                        const float* bias_v,
                                                        __bf16* out0, __bf16* out1, float* outf,
                                                        int M, int N, int K) {
  __shared__ __align__(16) __bf16 sA[2][128 * 32];
  __shared__ __align__(16) __bf16 sB[2][128 * 32];
  int tid = threadIdx.x, lane = tid & 31, w = tid >> 5;
  int row0 = blockIdx.x * 128, col0 = blockIdx.y * 128;
  int mw = (w & 3) * 32, nw = (w >> 2) * 64;
  f32x8 acc[2][4];
#pragma unroll
  for (int i = 0; i < 2; ++i)
#pragma unroll
    for (int j = 0; j < 4; ++j) acc[i][j] = (f32x8){0.f,0.f,0.f,0.f,0.f,0.f,0.f,0.f};

  stage_gemm_tile(A, Bt, sA[0], sB[0], tid, row0, col0, M, N, K, 0);

  int nsteps = K / 32;
  for (int it = 0; it < nsteps; ++it) {
    wait_async_all();          // async writes into buffer (it&1) done (issued last iter)
    __syncthreads();           // all waves see the staged tile / done reading other buffer
    if (it + 1 < nsteps)
      stage_gemm_tile(A, Bt, sA[(it + 1) & 1], sB[(it + 1) & 1], tid, row0, col0, M, N, K,
                      32 * (it + 1));
    const __bf16* cA = sA[it & 1];
    const __bf16* cB = sB[it & 1];
    bf16x16 a0 = frag_a(&cA[mw * 32], 32, lane);
    bf16x16 a1 = frag_a(&cA[(mw + 16) * 32], 32, lane);
#pragma unroll
    for (int j = 0; j < 4; ++j) {
      bf16x16 bf = frag_b(&cB[(nw + 16 * j) * 32], 32, lane);
      acc[0][j] = wmma_bf16(a0, bf, acc[0][j]);
      acc[1][j] = wmma_bf16(a1, bf, acc[1][j]);
    }
  }

  int nn = lane & 15, hl = lane >> 4;
#pragma unroll
  for (int i = 0; i < 2; ++i) {
#pragma unroll
    for (int j = 0; j < 4; ++j) {
      int gcol = col0 + nw + 16 * j + nn;
      float b0 = bias0 ? bias0[gcol] : 0.f;
      float bu = bias_u ? bias_u[gcol] : 0.f;
      float bv = bias_v ? bias_v[gcol] : 0.f;
#pragma unroll
      for (int r = 0; r < 8; ++r) {
        int grow = row0 + mw + 16 * i + r + 8 * hl;
        if (grow < M) {
          float va = acc[i][j][r];
          if (out0) out0[(size_t)grow * N + gcol] = (__bf16)(va + b0 + bu);
          if (out1) out1[(size_t)grow * N + gcol] = (__bf16)(va + b0 + bv);
          if (outf) outf[(size_t)grow * N + gcol] = va + b0;
        }
      }
    }
  }
}

// ---------------- fused rel-pos flash attention ----------------
// grid: (T/64, H, B); block 256 (8 waves). q fragments pinned in registers.
__global__ __launch_bounds__(256) void attn_kernel(const __bf16* __restrict__ qu,
                                                   const __bf16* __restrict__ qv,
                                                   const __bf16* __restrict__ kg,
                                                   const __bf16* __restrict__ vg,
                                                   const __bf16* __restrict__ pg,
                                                   const int* __restrict__ xlens,
                                                   __bf16* __restrict__ ctx) {
  __shared__ __align__(16) __bf16 s_k[64 * 64];    // [s'][dk]
  __shared__ __align__(16) __bf16 s_vt[64 * 64];   // [dk][s']  (transposed v)
  __shared__ __align__(16) __bf16 s_p[128 * 64];   // [j'][dk]
  __shared__ __align__(16) __bf16 s_bd[64 * 128];  // raw q_v . p^T (pre-shift)
  __shared__ __align__(16) __bf16 s_sp[64 * 64];   // scores, then probs
  __shared__ float s_m[64], s_l[64], s_scale[64];

  int tid = threadIdx.x, lane = tid & 31, w = tid >> 5;
  int nn = lane & 15, hl = lane >> 4;
  int t0 = blockIdx.x * 64, head = blockIdx.y, b = blockIdx.z;
  size_t bT = (size_t)b * T_;
  int mt = (w & 3) * 16;       // query-row group of this wave
  int nh = w >> 2;             // 0/1: column half
  int xlen = xlens[b];

  // q fragments (rows t0+mt .. +15, dk 0..63) loaded once from global
  bf16x16 qu_f[2], qv_f[2];
#pragma unroll
  for (int ks = 0; ks < 2; ++ks) {
    size_t base = (bT + t0 + mt) * D_ + head * DK_ + 32 * ks;
    qu_f[ks] = frag_a(qu + base, D_, lane);
    qv_f[ks] = frag_a(qv + base, D_, lane);
  }

  f32x8 o[2];
  o[0] = (f32x8){0.f,0.f,0.f,0.f,0.f,0.f,0.f,0.f};
  o[1] = o[0];
  if (tid < 64) { s_m[tid] = -3.0e38f; s_l[tid] = 0.f; }
  __syncthreads();

  for (int s0 = 0; s0 < T_; s0 += 64) {
    if (s0 >= xlen) break;   // everything beyond is masked -> contributes 0
    __syncthreads();         // protect LDS from previous iteration's readers

    // ---- stage k (async), v^T (transposing scatter), p (async) ----
#pragma unroll
    for (int i = 0; i < 2; ++i) {
      int c = tid + 256 * i;
      int sr = c >> 3, dk0 = (c & 7) * 8;
      async_copy16(&kg[(bT + s0 + sr) * D_ + head * DK_ + dk0], &s_k[sr * 64 + dk0]);
      bf16x8 vv = *(const bf16x8*)&vg[(bT + s0 + sr) * D_ + head * DK_ + dk0];
#pragma unroll
      for (int e = 0; e < 8; ++e) s_vt[(dk0 + e) * 64 + sr] = vv[e];
    }
    int j0 = (T_ - 1) + s0 - t0 - 63;   // >= 0 always
#pragma unroll
    for (int i = 0; i < 4; ++i) {
      int c = tid + 256 * i;
      int jr = c >> 3, dk0 = (c & 7) * 8;
      int prow = j0 + jr; prow = prow > (2 * T_ - 2) ? (2 * T_ - 2) : prow;
      async_copy16(&pg[(size_t)prow * D_ + head * DK_ + dk0], &s_p[jr * 64 + dk0]);
    }
    wait_async_all();
    __syncthreads();

    // ---- bd = q_v . p^T  (64 x 128 raw), ac = q_u . k^T (64 x 64) ----
    f32x8 dacc[4];
#pragma unroll
    for (int j = 0; j < 4; ++j) dacc[j] = (f32x8){0.f,0.f,0.f,0.f,0.f,0.f,0.f,0.f};
    f32x8 cac[2];
    cac[0] = dacc[0]; cac[1] = dacc[0];
#pragma unroll
    for (int ks = 0; ks < 2; ++ks) {
#pragma unroll
      for (int j = 0; j < 4; ++j) {
        bf16x16 bf = frag_b(&s_p[(nh * 64 + 16 * j) * 64 + 32 * ks], 64, lane);
        dacc[j] = wmma_bf16(qv_f[ks], bf, dacc[j]);
      }
#pragma unroll
      for (int j = 0; j < 2; ++j) {
        bf16x16 bf = frag_b(&s_k[(nh * 32 + 16 * j) * 64 + 32 * ks], 64, lane);
        cac[j] = wmma_bf16(qu_f[ks], bf, cac[j]);
      }
    }
#pragma unroll
    for (int j = 0; j < 4; ++j)
#pragma unroll
      for (int r = 0; r < 8; ++r) {
        int R = mt + r + 8 * hl, Cc = nh * 64 + 16 * j + nn;
        s_bd[R * 128 + Cc] = (__bf16)dacc[j][r];
      }
    __syncthreads();

    // ---- combine with Transformer-XL shift: score[t,s] uses bd[t, 63+s-t] ----
#pragma unroll
    for (int j = 0; j < 2; ++j)
#pragma unroll
      for (int r = 0; r < 8; ++r) {
        int R = mt + r + 8 * hl, Cc = nh * 32 + 16 * j + nn;
        float sv = (cac[j][r] + (float)s_bd[R * 128 + 63 + Cc - R]) * 0.125f;
        s_sp[R * 64 + Cc] = (__bf16)sv;
      }
    __syncthreads();

    // ---- online softmax (one thread per query row) ----
    if (tid < 64) {
      int r = tid;
      float mold = s_m[r], mnew = mold;
      for (int sc = 0; sc < 64; ++sc)
        if (s0 + sc < xlen) mnew = fmaxf(mnew, (float)s_sp[r * 64 + sc]);
      float scale = __expf(mold - mnew);
      float l = s_l[r] * scale;
      for (int sc = 0; sc < 64; ++sc) {
        float pv = 0.f;
        if (s0 + sc < xlen) pv = __expf((float)s_sp[r * 64 + sc] - mnew);
        s_sp[r * 64 + sc] = (__bf16)pv;
        l += pv;
      }
      s_m[r] = mnew; s_l[r] = l; s_scale[r] = scale;
    }
    __syncthreads();

    // ---- rescale O and accumulate P @ V ----
    float scl[8];
#pragma unroll
    for (int r = 0; r < 8; ++r) scl[r] = s_scale[mt + r + 8 * hl];
#pragma unroll
    for (int j = 0; j < 2; ++j)
#pragma unroll
      for (int r = 0; r < 8; ++r) o[j][r] *= scl[r];
#pragma unroll
    for (int ks = 0; ks < 2; ++ks) {
      bf16x16 a = frag_a(&s_sp[mt * 64 + 32 * ks], 64, lane);
#pragma unroll
      for (int j = 0; j < 2; ++j) {
        bf16x16 bf = frag_b(&s_vt[(nh * 32 + 16 * j) * 64 + 32 * ks], 64, lane);
        o[j] = wmma_bf16(a, bf, o[j]);
      }
    }
  }

  // ---- normalize and write context (bf16, [b*T+t][h*64+dk]) ----
#pragma unroll
  for (int r = 0; r < 8; ++r) {
    int R = mt + r + 8 * hl;
    float li = s_l[R];
    float inv = li > 0.f ? 1.f / li : 0.f;
#pragma unroll
    for (int j = 0; j < 2; ++j) {
      int col = nh * 32 + 16 * j + nn;
      ctx[(bT + t0 + R) * D_ + head * DK_ + col] = (__bf16)(o[j][r] * inv);
    }
  }
}

// ---------------- host launch ----------------
extern "C" void kernel_launch(void* const* d_in, const int* in_sizes, int n_in,
                              void* d_out, int out_size, void* d_ws, size_t ws_size,
                              hipStream_t stream) {
  const float* x       = (const float*)d_in[0];
  const float* pos_emb = (const float*)d_in[2];
  const float* ln_g    = (const float*)d_in[3];
  const float* ln_b    = (const float*)d_in[4];
  const float* Wq = (const float*)d_in[5];  const float* bq = (const float*)d_in[6];
  const float* Wk = (const float*)d_in[7];  const float* bk = (const float*)d_in[8];
  const float* Wv = (const float*)d_in[9];  const float* bv = (const float*)d_in[10];
  const float* Wpos = (const float*)d_in[11];
  const float* pbu  = (const float*)d_in[12];
  const float* pbv  = (const float*)d_in[13];
  const float* Wo = (const float*)d_in[14]; const float* bo = (const float*)d_in[15];
  const int* xlens = (const int*)d_in[1];

  char* ws = (char*)d_ws;
  size_t off = 0;
  auto take = [&](size_t bytes) -> void* {
    void* p = ws + off;
    off += (bytes + 255) & ~(size_t)255;
    return p;
  };
  const size_t actB = (size_t)B_ * T_ * D_ * sizeof(__bf16);
  const size_t wgtB = (size_t)D_ * D_ * sizeof(__bf16);
  const size_t posB = (size_t)(2 * T_ - 1) * D_ * sizeof(__bf16);

  __bf16* xn   = (__bf16*)take(actB);
  __bf16* Wqt  = (__bf16*)take(wgtB);
  __bf16* Wkt  = (__bf16*)take(wgtB);
  __bf16* Wvt  = (__bf16*)take(wgtB);
  __bf16* Wpt  = (__bf16*)take(wgtB);
  __bf16* Wot  = (__bf16*)take(wgtB);
  __bf16* posb = (__bf16*)take(posB);
  __bf16* q_u  = (__bf16*)take(actB);
  __bf16* q_v  = (__bf16*)take(actB);
  __bf16* kbuf = (__bf16*)take(actB);
  __bf16* vbuf = (__bf16*)take(actB);
  __bf16* pbuf = (__bf16*)take(posB);
  __bf16* ctx  = (__bf16*)take(actB);

  const int M = B_ * T_;          // 8192
  const int Mp = 2 * T_ - 1;      // 2047

  ln_kernel<<<M / 8, 256, 0, stream>>>(x, ln_g, ln_b, xn);

  int wblk = (D_ * D_ + 255) / 256;
  transpose_bf16_kernel<<<wblk, 256, 0, stream>>>(Wq, Wqt, D_, D_);
  transpose_bf16_kernel<<<wblk, 256, 0, stream>>>(Wk, Wkt, D_, D_);
  transpose_bf16_kernel<<<wblk, 256, 0, stream>>>(Wv, Wvt, D_, D_);
  transpose_bf16_kernel<<<wblk, 256, 0, stream>>>(Wpos, Wpt, D_, D_);
  transpose_bf16_kernel<<<wblk, 256, 0, stream>>>(Wo, Wot, D_, D_);
  convert_bf16_kernel<<<(Mp * D_ + 255) / 256, 256, 0, stream>>>(pos_emb, posb, Mp * D_);

  dim3 gA(M / 128, D_ / 128);
  dim3 gP((Mp + 127) / 128, D_ / 128);
  // q (two outputs: +pos_bias_u / +pos_bias_v), k, v, p
  gemm_bf16_kernel<<<gA, 256, 0, stream>>>(xn, Wqt, bq, pbu, pbv, q_u, q_v, nullptr, M, D_, D_);
  gemm_bf16_kernel<<<gA, 256, 0, stream>>>(xn, Wkt, bk, nullptr, nullptr, kbuf, nullptr, nullptr, M, D_, D_);
  gemm_bf16_kernel<<<gA, 256, 0, stream>>>(xn, Wvt, bv, nullptr, nullptr, vbuf, nullptr, nullptr, M, D_, D_);
  gemm_bf16_kernel<<<gP, 256, 0, stream>>>(posb, Wpt, nullptr, nullptr, nullptr, pbuf, nullptr, nullptr, Mp, D_, D_);

  attn_kernel<<<dim3(T_ / 64, H_, B_), 256, 0, stream>>>(q_u, q_v, kbuf, vbuf, pbuf, xlens, ctx);

  gemm_bf16_kernel<<<gA, 256, 0, stream>>>(ctx, Wot, bo, nullptr, nullptr, nullptr, nullptr,
                                           (float*)d_out, M, D_, D_);

  // tuple output: append xlens (int bits) after the (B,T,D) fp32 tensor
  (void)hipMemcpyAsync((float*)d_out + (size_t)M * D_, d_in[1], B_ * sizeof(int),
                       hipMemcpyDeviceToDevice, stream);
}
